// SCOPFGNN_36137854828909
// MI455X (gfx1250) — compile-verified, matrix operand order checked by reading, not run on hardware
//
#include <hip/hip_runtime.h>
#include <hip/hip_bf16.h>
#include <cstdint>
#include <cstddef>

typedef __attribute__((ext_vector_type(16))) __bf16 v16bf;
typedef __attribute__((ext_vector_type(8)))  __bf16 v8bf;
typedef __attribute__((ext_vector_type(8)))  float  v8f;

static constexpr int NN = 100000;   // nodes
static constexpr int NE = 200000;   // edges
static constexpr int H  = 32;
static constexpr int G  = 100;
static constexpr int NL = 4;
static constexpr int NGRAPH = 50;
static constexpr int NCHUNK = 65;   // 64 chunks of t[:,j] (x) h  +  1 chunk for b2 term

// ---------------------------------------------------------------------------
// Node embedding: h = relu(x @ w_embed + b_embed)   (N x 6) @ (6 x 32)
// ---------------------------------------------------------------------------
__global__ void embed_kernel(const float* __restrict__ x,
                             const float* __restrict__ w,
                             const float* __restrict__ b,
                             float* __restrict__ h) {
  int idx = blockIdx.x * blockDim.x + threadIdx.x;   // exactly NN*32 threads
  int n = idx >> 5, c = idx & 31;
  float s = b[c];
#pragma unroll
  for (int i = 0; i < 6; ++i) s += x[n * 6 + i] * w[i * 32 + c];
  h[idx] = fmaxf(s, 0.0f);
}

// ---------------------------------------------------------------------------
// BatchNorm batch statistics: acc[c]=sum, acc[32+c]=sumsq
// ---------------------------------------------------------------------------
__global__ void bn_stats_kernel(const float* __restrict__ h, float* __restrict__ acc) {
  __shared__ float ss[64];
  if (threadIdx.x < 64) ss[threadIdx.x] = 0.0f;
  __syncthreads();
  int c = threadIdx.x & 31;
  int row0 = blockIdx.x * (blockDim.x >> 5) + (threadIdx.x >> 5);
  int rstride = gridDim.x * (blockDim.x >> 5);
  float s = 0.0f, q = 0.0f;
  for (int n = row0; n < NN; n += rstride) {
    float v = h[(size_t)n * 32 + c];
    s += v; q += v * v;
  }
  atomicAdd(&ss[c], s);
  atomicAdd(&ss[32 + c], q);
  __syncthreads();
  if (threadIdx.x < 64) atomicAdd(&acc[threadIdx.x], ss[threadIdx.x]);
}

__global__ void bn_apply_kernel(float* __restrict__ h,
                                const float* __restrict__ acc,
                                const float* __restrict__ gamma,
                                const float* __restrict__ beta) {
  int idx = blockIdx.x * blockDim.x + threadIdx.x;   // NN*32 threads
  int c = idx & 31;
  float mu  = acc[c] * (1.0f / NN);
  float var = acc[32 + c] * (1.0f / NN) - mu * mu;
  h[idx] = (h[idx] - mu) * rsqrtf(var + 1e-5f) * gamma[c] + beta[c];
}

// ---------------------------------------------------------------------------
// In-degree -> inv_deg = 1/max(deg,1)
// ---------------------------------------------------------------------------
__global__ void deg_kernel(const int* __restrict__ dst, float* __restrict__ deg) {
  int e = blockIdx.x * blockDim.x + threadIdx.x;
  if (e < NE) atomicAdd(&deg[dst[e]], 1.0f);
}
__global__ void invdeg_kernel(float* __restrict__ deg) {
  int n = blockIdx.x * blockDim.x + threadIdx.x;
  if (n < NN) deg[n] = 1.0f / fmaxf(deg[n], 1.0f);
}

// ---------------------------------------------------------------------------
// Edge MLP hidden layer: t = relu(edge_attr @ w1 + b1) -> bf16   (E x 64)
// ---------------------------------------------------------------------------
__global__ void edge_mlp_kernel(const float* __restrict__ ea,
                                const float* __restrict__ w1,  // 4 x 64 (layer base)
                                const float* __restrict__ b1,  // 64
                                __bf16* __restrict__ t) {
  int idx = blockIdx.x * blockDim.x + threadIdx.x;   // exactly NE*64 threads
  int e = idx >> 6, k = idx & 63;
  float s = b1[k];
#pragma unroll
  for (int i = 0; i < 4; ++i) s += ea[e * 4 + i] * w1[i * 64 + k];
  t[idx] = (__bf16)fmaxf(s, 0.0f);
}

// ---------------------------------------------------------------------------
// Message GEMM via WMMA:  msg(32e x 32) = A(32e x 2080) x B(2080 x 32)
//   chunk j<64 : A = t[:,j] (outer) h[src]   B = w2[j] as 32x32
//   chunk 64   : A = h[src]                  B = b2 as 32x32
// One wave handles a 32-edge supertile (two 16-row A fragments sharing the
// same B fragments -> 4 WMMAs per chunk, LDS traffic halved per edge).
// Scatter-add msg into agg[dst].
// B staged in LDS in WMMA-fragment order (bf16, 130 KB, fits 320KB WGP LDS).
// ---------------------------------------------------------------------------
__global__ void msg_kernel(const float*  __restrict__ h,
                           const __bf16* __restrict__ tbuf,
                           const float*  __restrict__ w2,   // layer base: 64*1024
                           const float*  __restrict__ b2,   // layer base: 1024
                           const int*    __restrict__ src,
                           const int*    __restrict__ dst,
                           float*        __restrict__ agg) {
  extern __shared__ char smem[];
  __bf16* ldsB = (__bf16*)smem;

  // Cooperative fill, coalesced on the global side: flat f = j*1024 + k*32 + o
  // (consecutive threads read consecutive w2 floats), scattered into LDS in
  // WMMA-B fragment order: ldsB[((j*2+tile)*32 + lane)*16 + p]
  for (int f = threadIdx.x; f < NCHUNK * 1024; f += blockDim.x) {
    int j   = f >> 10;
    int rem = f & 1023;
    int k   = rem >> 5;
    int o   = rem & 31;
    int tile = o >> 4, pp = o & 15;
    int q = (k >> 3) & 1;
    int p = (k & 7) + ((k >= 16) ? 8 : 0);
    int lane = q * 16 + pp;
    float v = (j < 64) ? w2[(size_t)j * 1024 + k * 32 + o] : b2[k * 32 + o];
    ldsB[((size_t)(j * 2 + tile) * 32 + lane) * 16 + p] = (__bf16)v;
  }
  __syncthreads();

  const int lane = threadIdx.x & 31;
  const int q = lane >> 4, pp = lane & 15;
  int wid = (blockIdx.x * blockDim.x + threadIdx.x) >> 5;
  int nw  = (gridDim.x * blockDim.x) >> 5;
  const int nst = NE / 32;   // 6250 supertiles, exact

  for (int st = wid; st < nst; st += nw) {
    int e0 = st * 32 + pp;        // lane's row in A-fragment 0 (edges 0..15)
    int e1 = st * 32 + 16 + pp;   // lane's row in A-fragment 1 (edges 16..31)
    int s0 = src[e0];
    int s1 = src[e1];
    const float* hr0 = h + (size_t)s0 * H;
    const float* hr1 = h + (size_t)s1 * H;

    // Gather h[src] and convert to bf16 once per supertile.
    v16bf hbf0, hbf1;
#pragma unroll
    for (int u = 0; u < 8; ++u) {
      hbf0[u]     = (__bf16)hr0[q * 8 + u];        // K 0-7   / 8-15
      hbf0[8 + u] = (__bf16)hr0[16 + q * 8 + u];   // K 16-23 / 24-31
      hbf1[u]     = (__bf16)hr1[q * 8 + u];
      hbf1[8 + u] = (__bf16)hr1[16 + q * 8 + u];
    }

    v8f c00 = {0, 0, 0, 0, 0, 0, 0, 0};   // A0 x Btile0
    v8f c01 = {0, 0, 0, 0, 0, 0, 0, 0};   // A0 x Btile1
    v8f c10 = {0, 0, 0, 0, 0, 0, 0, 0};   // A1 x Btile0
    v8f c11 = {0, 0, 0, 0, 0, 0, 0, 0};   // A1 x Btile1
    const v8bf* tp0 = (const v8bf*)(tbuf + (size_t)e0 * 64);
    const v8bf* tp1 = (const v8bf*)(tbuf + (size_t)e1 * 64);

    // 64 outer-product chunks, grouped 8 per packed t-load (b128), unrolled so
    // the scheduler overlaps ds_load_b128 / bf16 A-build / WMMA across chunks.
#pragma unroll 1
    for (int jg = 0; jg < 8; ++jg) {
      v8bf tv0 = tp0[jg];
      v8bf tv1 = tp1[jg];
#pragma unroll
      for (int jj = 0; jj < 8; ++jj) {
        int j = jg * 8 + jj;
        v16bf a0 = hbf0 * tv0[jj];   // native bf16 packed mul
        v16bf a1 = hbf1 * tv1[jj];
        const v16bf b0 = *(const v16bf*)(ldsB + ((size_t)(j * 2 + 0) * 32 + lane) * 16);
        const v16bf b1 = *(const v16bf*)(ldsB + ((size_t)(j * 2 + 1) * 32 + lane) * 16);
        c00 = __builtin_amdgcn_wmma_f32_16x16x32_bf16(false, a0, false, b0, (short)0, c00, false, false);
        c01 = __builtin_amdgcn_wmma_f32_16x16x32_bf16(false, a0, false, b1, (short)0, c01, false, false);
        c10 = __builtin_amdgcn_wmma_f32_16x16x32_bf16(false, a1, false, b0, (short)0, c10, false, false);
        c11 = __builtin_amdgcn_wmma_f32_16x16x32_bf16(false, a1, false, b1, (short)0, c11, false, false);
      }
    }
    // b2 chunk (j == 64): A = h[src]
    {
      const v16bf b0 = *(const v16bf*)(ldsB + ((size_t)(64 * 2 + 0) * 32 + lane) * 16);
      const v16bf b1 = *(const v16bf*)(ldsB + ((size_t)(64 * 2 + 1) * 32 + lane) * 16);
      c00 = __builtin_amdgcn_wmma_f32_16x16x32_bf16(false, hbf0, false, b0, (short)0, c00, false, false);
      c01 = __builtin_amdgcn_wmma_f32_16x16x32_bf16(false, hbf0, false, b1, (short)0, c01, false, false);
      c10 = __builtin_amdgcn_wmma_f32_16x16x32_bf16(false, hbf1, false, b0, (short)0, c10, false, false);
      c11 = __builtin_amdgcn_wmma_f32_16x16x32_bf16(false, hbf1, false, b1, (short)0, c11, false, false);
    }

    // C layout: lanes 0-15 rows 0-7, lanes 16-31 rows 8-15; col = lane&15
#pragma unroll
    for (int r = 0; r < 8; ++r) {
      int er0 = st * 32 + q * 8 + r;        // fragment-0 rows
      int er1 = st * 32 + 16 + q * 8 + r;   // fragment-1 rows
      int d0  = dst[er0];
      int d1  = dst[er1];
      atomicAdd(agg + (size_t)d0 * H + pp,      c00[r]);
      atomicAdd(agg + (size_t)d0 * H + 16 + pp, c01[r]);
      atomicAdd(agg + (size_t)d1 * H + pp,      c10[r]);
      atomicAdd(agg + (size_t)d1 * H + 16 + pp, c11[r]);
    }
  }
}

// ---------------------------------------------------------------------------
// Node update via WMMA: h' = relu(agg*inv_deg + h @ root + bias)
// ---------------------------------------------------------------------------
__global__ void node_update_kernel(const float* __restrict__ h,
                                   const float* __restrict__ agg,
                                   const float* __restrict__ inv_deg,
                                   const float* __restrict__ root,  // 32x32 layer base
                                   const float* __restrict__ bias,  // 32
                                   float* __restrict__ hout) {
  const int lane = threadIdx.x & 31;
  const int q = lane >> 4, pp = lane & 15;

  v16bf b0, b1;
#pragma unroll
  for (int p = 0; p < 16; ++p) {
    int k = (p < 8) ? (q * 8 + p) : (16 + q * 8 + (p - 8));
    b0[p] = (__bf16)root[k * 32 + pp];
    b1[p] = (__bf16)root[k * 32 + 16 + pp];
  }
  const float bias0 = bias[pp], bias1 = bias[16 + pp];

  int wid = (blockIdx.x * blockDim.x + threadIdx.x) >> 5;
  int nw  = (gridDim.x * blockDim.x) >> 5;
  const int ntiles = NN / 16;   // 6250, exact

  for (int tile = wid; tile < ntiles; tile += nw) {
    int n = tile * 16 + pp;
    const float* hrow = h + (size_t)n * H;
    v16bf a;
#pragma unroll
    for (int u = 0; u < 8; ++u) {
      a[u]     = (__bf16)hrow[q * 8 + u];
      a[8 + u] = (__bf16)hrow[16 + q * 8 + u];
    }

    v8f c0 = {0, 0, 0, 0, 0, 0, 0, 0};
    v8f c1 = {0, 0, 0, 0, 0, 0, 0, 0};
    c0 = __builtin_amdgcn_wmma_f32_16x16x32_bf16(false, a, false, b0, (short)0, c0, false, false);
    c1 = __builtin_amdgcn_wmma_f32_16x16x32_bf16(false, a, false, b1, (short)0, c1, false, false);

#pragma unroll
    for (int r = 0; r < 8; ++r) {
      int nd = tile * 16 + q * 8 + r;
      float id = inv_deg[nd];
      float v0 = c0[r] + agg[(size_t)nd * H + pp] * id + bias0;
      float v1 = c1[r] + agg[(size_t)nd * H + 16 + pp] * id + bias1;
      hout[(size_t)nd * H + pp]      = fmaxf(v0, 0.0f);
      hout[(size_t)nd * H + 16 + pp] = fmaxf(v1, 0.0f);
    }
  }
}

// ---------------------------------------------------------------------------
// Global mean pool (accumulate) + heads
// ---------------------------------------------------------------------------
__global__ void pool_kernel(const float* __restrict__ h, const int* __restrict__ batch,
                            float* __restrict__ hg, float* __restrict__ cnt) {
  int idx = blockIdx.x * blockDim.x + threadIdx.x;   // NN*32 threads
  int n = idx >> 5, c = idx & 31;
  int b = batch[n];
  atomicAdd(&hg[b * 32 + c], h[idx]);
  if (c == 0) atomicAdd(&cnt[b], 1.0f);
}

__global__ void heads_kernel(const float* __restrict__ hg, const float* __restrict__ cnt,
                             const float* __restrict__ wkw1, const float* __restrict__ wkb1,
                             const float* __restrict__ wkw2, const float* __restrict__ wkb2,
                             const float* __restrict__ ujw1, const float* __restrict__ ujb1,
                             const float* __restrict__ ujw2, const float* __restrict__ ujb2,
                             const float* __restrict__ zkw1, const float* __restrict__ zkb1,
                             const float* __restrict__ zkw2, const float* __restrict__ zkb2,
                             float* __restrict__ out) {
  __shared__ float hgm[32];
  __shared__ float hid[96];   // [0,32): wk  [32,64): uj  [64,96): zk
  int g = blockIdx.x;
  int tid = threadIdx.x;
  if (tid < 32) hgm[tid] = hg[g * 32 + tid] / fmaxf(cnt[g], 1.0f);
  __syncthreads();
  if (tid < 96) {
    int head = tid >> 5, c = tid & 31;
    const float* w1 = (head == 0) ? wkw1 : (head == 1) ? ujw1 : zkw1;
    const float* b1 = (head == 0) ? wkb1 : (head == 1) ? ujb1 : zkb1;
    float s = b1[c];
#pragma unroll
    for (int i = 0; i < 32; ++i) s += hgm[i] * w1[i * 32 + c];
    hid[tid] = fmaxf(s, 0.0f);
  }
  __syncthreads();
  if (tid == 0) {
    float s = wkb2[0];
#pragma unroll
    for (int i = 0; i < 32; ++i) s += hid[i] * wkw2[i];
    out[g] = s;
  }
  if (tid < G) {
    float su = ujb2[tid], sz = zkb2[tid];
#pragma unroll
    for (int i = 0; i < 32; ++i) {
      su += hid[32 + i] * ujw2[i * G + tid];
      sz += hid[64 + i] * zkw2[i * G + tid];
    }
    out[NGRAPH + (size_t)g * G + tid] = su;
    out[NGRAPH + (size_t)NGRAPH * G + (size_t)g * G + tid] = sz;
  }
}

// ---------------------------------------------------------------------------
extern "C" void kernel_launch(void* const* d_in, const int* in_sizes, int n_in,
                              void* d_out, int out_size, void* d_ws, size_t ws_size,
                              hipStream_t stream) {
  (void)in_sizes; (void)n_in; (void)out_size; (void)ws_size;

  const float* x         = (const float*)d_in[0];
  const int*   eidx      = (const int*)d_in[1];
  const float* eattr     = (const float*)d_in[2];
  const int*   batch     = (const int*)d_in[3];
  const float* w_embed   = (const float*)d_in[4];
  const float* b_embed   = (const float*)d_in[5];
  const float* bn_gamma  = (const float*)d_in[6];
  const float* bn_beta   = (const float*)d_in[7];
  const float* conv_w1   = (const float*)d_in[8];
  const float* conv_b1   = (const float*)d_in[9];
  const float* conv_w2   = (const float*)d_in[10];
  const float* conv_b2   = (const float*)d_in[11];
  const float* conv_root = (const float*)d_in[12];
  const float* conv_bias = (const float*)d_in[13];
  const float* wk_w1 = (const float*)d_in[14];
  const float* wk_b1 = (const float*)d_in[15];
  const float* wk_w2 = (const float*)d_in[16];
  const float* wk_b2 = (const float*)d_in[17];
  const float* uj_w1 = (const float*)d_in[18];
  const float* uj_b1 = (const float*)d_in[19];
  const float* uj_w2 = (const float*)d_in[20];
  const float* uj_b2 = (const float*)d_in[21];
  const float* zk_w1 = (const float*)d_in[22];
  const float* zk_b1 = (const float*)d_in[23];
  const float* zk_w2 = (const float*)d_in[24];
  const float* zk_b2 = (const float*)d_in[25];
  float* out = (float*)d_out;

  char* ws = (char*)d_ws;
  size_t off = 0;
  auto alloc = [&](size_t bytes) -> void* {
    void* p = ws + off;
    off = (off + bytes + 255) & ~(size_t)255;
    return p;
  };
  float*  h0    = (float*)alloc((size_t)NN * H * 4);
  float*  h1    = (float*)alloc((size_t)NN * H * 4);
  float*  agg   = (float*)alloc((size_t)NN * H * 4);
  __bf16* tb    = (__bf16*)alloc((size_t)NE * 64 * 2);
  float*  invd  = (float*)alloc((size_t)NN * 4);
  float*  bnacc = (float*)alloc(64 * 4);
  float*  hg    = (float*)alloc((size_t)NGRAPH * H * 4);
  float*  cnt   = (float*)alloc((size_t)NGRAPH * 4);

  const int* src = eidx;
  const int* dst = eidx + NE;

  // embed + BN
  embed_kernel<<<NN * H / 256, 256, 0, stream>>>(x, w_embed, b_embed, h0);
  hipMemsetAsync(bnacc, 0, 64 * 4, stream);
  bn_stats_kernel<<<256, 256, 0, stream>>>(h0, bnacc);
  bn_apply_kernel<<<NN * H / 256, 256, 0, stream>>>(h0, bnacc, bn_gamma, bn_beta);

  // degrees
  hipMemsetAsync(invd, 0, (size_t)NN * 4, stream);
  deg_kernel<<<(NE + 255) / 256, 256, 0, stream>>>(dst, invd);
  invdeg_kernel<<<(NN + 255) / 256, 256, 0, stream>>>(invd);

  float* hcur = h0;
  float* hnxt = h1;
  const size_t ldsBytes = (size_t)NCHUNK * 2 * 32 * 16 * sizeof(__bf16);  // 133120 B

  for (int l = 0; l < NL; ++l) {
    edge_mlp_kernel<<<NE * 64 / 256, 256, 0, stream>>>(
        eattr, conv_w1 + (size_t)l * 4 * 64, conv_b1 + (size_t)l * 64, tb);
    hipMemsetAsync(agg, 0, (size_t)NN * H * 4, stream);
    msg_kernel<<<384, 256, ldsBytes, stream>>>(
        hcur, tb, conv_w2 + (size_t)l * 64 * 1024, conv_b2 + (size_t)l * 1024,
        src, dst, agg);
    node_update_kernel<<<256, 256, 0, stream>>>(
        hcur, agg, invd, conv_root + (size_t)l * H * H, conv_bias + (size_t)l * H, hnxt);
    float* t = hcur; hcur = hnxt; hnxt = t;
  }

  // pool + heads
  hipMemsetAsync(hg, 0, (size_t)NGRAPH * H * 4, stream);
  hipMemsetAsync(cnt, 0, (size_t)NGRAPH * 4, stream);
  pool_kernel<<<NN * H / 256, 256, 0, stream>>>(hcur, batch, hg, cnt);
  heads_kernel<<<NGRAPH, 128, 0, stream>>>(hg, cnt,
      wk_w1, wk_b1, wk_w2, wk_b2,
      uj_w1, uj_b1, uj_w2, uj_b2,
      zk_w1, zk_b1, zk_w2, zk_b2, out);
}
